// SkipGramModel_47742856462411
// MI455X (gfx1250) — matrix operand order repaired and checked by reading, not compiled, not run on hardware
//
#include <hip/hip_runtime.h>

typedef __attribute__((ext_vector_type(16))) _Float16 v16h;
typedef __attribute__((ext_vector_type(8)))  float    v8f;

#define D_DIM 128
#define K_NEG 5
#define WAVES_PER_BLOCK 32
#define BLOCK_THREADS (WAVES_PER_BLOCK * 32)

// One wave (32 lanes) handles one batch element b.
// A (16x128 f16): rows 0..5 = [v_pos ; v_neg0..4], rows 6..15 duplicate row 0.
// B (128x16 f16): every column = u[b]  (so every D column holds the scores).
// D = A*B (4x v_wmma_f32_16x16x32_f16): lane 0 holds scores in acc[0..5].
__global__ __launch_bounds__(BLOCK_THREADS)
void sg_loss_kernel(const int* __restrict__ pos_u,
                    const int* __restrict__ pos_v,
                    const int* __restrict__ neg_v,
                    const unsigned char* __restrict__ frozen_mask,
                    const float* __restrict__ u_emb,
                    const float* __restrict__ frozen_emb,
                    const float* __restrict__ v_emb,
                    float* __restrict__ block_part,
                    int B)
{
    __shared__ float wave_part[WAVES_PER_BLOCK];

    const int tid  = threadIdx.x;
    const int wave = tid >> 5;
    const int lane = tid & 31;
    const int b    = blockIdx.x * WAVES_PER_BLOCK + wave;

    float loss = 0.0f;

    if (b < B) {
        const int pu = pos_u[b];
        const int pv = pos_v[b];
        const unsigned char fm = frozen_mask[pu];
        const float* urow = (fm ? frozen_emb : u_emb) + (size_t)pu * D_DIM;

        // Select this lane's A-matrix row pointer (row index m = lane % 16).
        const int m = lane & 15;
        const float* arow;
        if (m >= 1 && m <= K_NEG) {
            arow = v_emb + (size_t)neg_v[b * K_NEG + (m - 1)] * D_DIM;
        } else {
            arow = v_emb + (size_t)pv * D_DIM;   // row 0 (and dup for rows 6..15)
        }

        const int hsel = (lane >> 4) << 3;       // 0 for lanes 0-15, 8 for lanes 16-31

        v8f acc = {};
        #pragma unroll
        for (int c = 0; c < 4; ++c) {
            const int kbase = 32 * c;

            // B fragment: every column = u; element h <- K = kbase + hsel + (h<8 ? h : h+8).
            v16h bfrag;
            #pragma unroll
            for (int h = 0; h < 16; ++h) bfrag[h] = (_Float16)urow[kbase + hsel + (h < 8 ? h : h + 8)];

            // A fragment (16-bit A 16x32 layout):
            //   halves 0..7  <- K = kbase + hsel + 0..7
            //   halves 8..15 <- K = kbase + hsel + 16..23
            const float4 f0 = *(const float4*)(arow + kbase + hsel);
            const float4 f1 = *(const float4*)(arow + kbase + hsel + 4);
            const float4 f2 = *(const float4*)(arow + kbase + hsel + 16);
            const float4 f3 = *(const float4*)(arow + kbase + hsel + 20);

            v16h afrag;
            afrag[0]  = (_Float16)f0.x; afrag[1]  = (_Float16)f0.y;
            afrag[2]  = (_Float16)f0.z; afrag[3]  = (_Float16)f0.w;
            afrag[4]  = (_Float16)f1.x; afrag[5]  = (_Float16)f1.y;
            afrag[6]  = (_Float16)f1.z; afrag[7]  = (_Float16)f1.w;
            afrag[8]  = (_Float16)f2.x; afrag[9]  = (_Float16)f2.y;
            afrag[10] = (_Float16)f2.z; afrag[11] = (_Float16)f2.w;
            afrag[12] = (_Float16)f3.x; afrag[13] = (_Float16)f3.y;
            afrag[14] = (_Float16)f3.z; afrag[15] = (_Float16)f3.w;

            // D(f32) = A(f16) x B(f16) + C
            acc = __builtin_amdgcn_wmma_f32_16x16x32_f16(
                false, afrag, false, bfrag, (short)0, acc, false, false);
        }

        if (lane == 0) {
            // positive score: -log_sigmoid(clip(s)) = softplus(-clip(s))
            float s0 = fminf(10.0f, fmaxf(-10.0f, acc[0]));
            float l  = log1pf(expf(-s0));
            // negatives: -log_sigmoid(-clip(s)) = softplus(clip(s))
            #pragma unroll
            for (int k = 1; k <= K_NEG; ++k) {
                float t = fminf(10.0f, fmaxf(-10.0f, acc[k]));
                l += log1pf(expf(t));
            }
            loss = l;
        }
    }

    if (lane == 0) wave_part[wave] = loss;
    __syncthreads();

    if (tid == 0) {
        float s = 0.0f;
        #pragma unroll
        for (int w = 0; w < WAVES_PER_BLOCK; ++w) s += wave_part[w];
        block_part[blockIdx.x] = s;
    }
}

// Deterministic final reduction of per-block partials, scaled by 1/B.
__global__ __launch_bounds__(1024)
void sg_reduce_kernel(const float* __restrict__ part, int n, float inv_b,
                      float* __restrict__ out)
{
    __shared__ float sm[1024];
    const int tid = threadIdx.x;
    float s = 0.0f;
    for (int i = tid; i < n; i += 1024) s += part[i];
    sm[tid] = s;
    __syncthreads();
    for (int stride = 512; stride > 0; stride >>= 1) {
        if (tid < stride) sm[tid] += sm[tid + stride];
        __syncthreads();
    }
    if (tid == 0) out[0] = sm[0] * inv_b;
}

extern "C" void kernel_launch(void* const* d_in, const int* in_sizes, int n_in,
                              void* d_out, int out_size, void* d_ws, size_t ws_size,
                              hipStream_t stream)
{
    (void)n_in; (void)out_size; (void)ws_size;
    const int*           pos_u  = (const int*)d_in[0];
    const int*           pos_v  = (const int*)d_in[1];
    const int*           neg_v  = (const int*)d_in[2];
    const unsigned char* fmask  = (const unsigned char*)d_in[3];  // jax bool -> 1 byte
    const float*         u_emb  = (const float*)d_in[4];
    const float*         f_emb  = (const float*)d_in[5];
    const float*         v_emb  = (const float*)d_in[6];
    float*               out    = (float*)d_out;
    float*               part   = (float*)d_ws;

    const int B    = in_sizes[0];
    const int nblk = (B + WAVES_PER_BLOCK - 1) / WAVES_PER_BLOCK;   // 4096 for B=131072

    sg_loss_kernel<<<nblk, BLOCK_THREADS, 0, stream>>>(
        pos_u, pos_v, neg_v, fmask, u_emb, f_emb, v_emb, part, B);
    sg_reduce_kernel<<<1, 1024, 0, stream>>>(part, nblk, 1.0f / (float)B, out);
}
